// MultiHeadAttention_3848290697596
// MI455X (gfx1250) — matrix operand order repaired and checked by reading, not compiled, not run on hardware
//
#include <hip/hip_runtime.h>
#include <hip/hip_bf16.h>
#include <stdint.h>

// ---------------- types ----------------
typedef __bf16 bf16_t;
typedef __attribute__((ext_vector_type(16))) __bf16 v16bf;
typedef __attribute__((ext_vector_type(8)))  __bf16 v8bf;
typedef __attribute__((ext_vector_type(8)))  float  v8f;

#define Bdim 4
#define Sdim 2048
#define HID  1024
#define Hh   16
#define Dd   64

// LDS tile geometry: 128 rows x 32 bf16 (64B) padded to 80B rows -> bank-conflict-free
#define LDSROW 40                      // elements per padded row (80 bytes)
#define TILE_ELEMS (128 * LDSROW)      // one 128x32 tile (padded)
#define STAGE_ELEMS (2 * TILE_ELEMS)   // A tile + B tile

__device__ __forceinline__ bf16_t to_bf16(float f) {
  union { float f; uint32_t u; } in; in.f = f;
  uint32_t u = in.u;
  u += 0x7FFFu + ((u >> 16) & 1u);          // round-to-nearest-even
  union { unsigned short s; bf16_t b; } out;
  out.s = (unsigned short)(u >> 16);
  return out.b;
}

// A fragment: 16x32 bf16, row-major source, lda in elements.
// lane 0-15 : row=lane,   K = {0..7, 16..23}
// lane 16-31: row=lane&15, K = {8..15, 24..31}
__device__ __forceinline__ v16bf load_frag_a(const bf16_t* base, int lda, int lane) {
  const int row = lane & 15;
  const int kh  = (lane >> 4) << 3;         // 0 or 8
  const bf16_t* p = base + (size_t)row * lda + kh;
  v8bf lo = *(const v8bf*)(p);
  v8bf hi = *(const v8bf*)(p + 16);
  v16bf a;
#pragma unroll
  for (int i = 0; i < 8; ++i) { a[i] = lo[i]; a[i + 8] = hi[i]; }
  return a;
}

// B fragment: 32x16 bf16 with SOURCE stored transposed (Bt[n,k], ldb elems).
// lane 0-15 : col=lane,   K = {0..7, 16..23} ; lane 16-31: col=lane&15, K = {8..15, 24..31}
__device__ __forceinline__ v16bf load_frag_b(const bf16_t* bt, int ldb, int lane) {
  const int col = lane & 15;
  const int kh  = (lane >> 4) << 3;         // 0 or 8
  const bf16_t* p = bt + (size_t)col * ldb + kh;
  v8bf lo = *(const v8bf*)(p);
  v8bf hi = *(const v8bf*)(p + 16);
  v16bf b;
#pragma unroll
  for (int i = 0; i < 8; ++i) { b[i] = lo[i]; b[i + 8] = hi[i]; }
  return b;
}

__device__ __forceinline__ v8f wmma_bf16(v16bf a, v16bf b, v8f c) {
  return __builtin_amdgcn_wmma_f32_16x16x32_bf16(
      false, a, false, b, (short)0, c, false, false);
}

// ---- CDNA5 async global->LDS copy (ASYNCcnt-tracked), 16B per lane ----
__device__ __forceinline__ void async_b128_to_lds(bf16_t* lds_ptr, const bf16_t* gptr) {
  uint32_t lds_off = (uint32_t)(uintptr_t)lds_ptr;   // low 32 bits = WG-relative LDS byte offset
  asm volatile("global_load_async_to_lds_b128 %0, %1, off"
               :: "v"(lds_off), "v"((unsigned long long)(uintptr_t)gptr)
               : "memory");
}
__device__ __forceinline__ void wait_asynccnt0() {
  asm volatile("s_wait_asynccnt 0x0" ::: "memory");
}

// Issue one BK=32 stage: A tile 128x32 from X[m0..,k0..], B tile 128x32 from
// W[n0..,k0..]; 512 16B chunks per tile, 256 threads x 2 chunks each.
__device__ __forceinline__ void stage_copy(const bf16_t* __restrict__ X,
                                           const bf16_t* __restrict__ W,
                                           int m0, int n0, int k0,
                                           bf16_t* smem, int stage, int tid) {
  bf16_t* sA = smem + stage * STAGE_ELEMS;
  bf16_t* sB = sA + TILE_ELEMS;
#pragma unroll
  for (int i = 0; i < 2; ++i) {
    const int c   = tid + i * 256;        // 0..511
    const int row = c >> 2;               // 0..127
    const int ch  = (c & 3) * 8;          // element offset of 16B chunk
    async_b128_to_lds(sA + row * LDSROW + ch,
                      X + (size_t)(m0 + row) * HID + k0 + ch);
    async_b128_to_lds(sB + row * LDSROW + ch,
                      W + (size_t)(n0 + row) * HID + k0 + ch);
  }
}

// Double-buffered LDS-staged GEMM core: block tile 128x128, 8 waves in 4x2,
// wave tile 32x64 -> 8 WMMAs per BK step from 6 LDS fragment loads.
__device__ __forceinline__ void gemm_core(const bf16_t* __restrict__ X,
                                          const bf16_t* __restrict__ W,
                                          int m0, int n0, bf16_t* smem,
                                          v8f acc[2][4]) {
  const int tid  = threadIdx.x;
  const int lane = tid & 31;
  const int wave = tid >> 5;
  const int wm = (wave >> 1) * 32;        // 0,32,64,96
  const int wn = (wave & 1) * 64;         // 0,64

  stage_copy(X, W, m0, n0, 0, smem, 0, tid);
  const int KT = HID / 32;
  for (int kt = 0; kt < KT; ++kt) {
    wait_asynccnt0();                     // this wave's current stage landed
    __syncthreads();                      // everyone's stage landed; prev buffer free
    if (kt + 1 < KT)
      stage_copy(X, W, m0, n0, (kt + 1) * 32, smem, (kt + 1) & 1, tid);

    const bf16_t* sA = smem + (kt & 1) * STAGE_ELEMS;
    const bf16_t* sB = sA + TILE_ELEMS;
    v16bf af0 = load_frag_a(sA + (wm + 0)  * LDSROW, LDSROW, lane);
    v16bf af1 = load_frag_a(sA + (wm + 16) * LDSROW, LDSROW, lane);
#pragma unroll
    for (int t = 0; t < 4; ++t) {
      v16bf bfr = load_frag_b(sB + (wn + t * 16) * LDSROW, LDSROW, lane);
      acc[0][t] = wmma_bf16(af0, bfr, acc[0][t]);
      acc[1][t] = wmma_bf16(af1, bfr, acc[1][t]);
    }
  }
}

// ---------------- f32 -> bf16 conversion ----------------
__global__ void __launch_bounds__(256)
cvt_bf16_kernel(const float* __restrict__ in, bf16_t* __restrict__ out, int n4) {
  int i = blockIdx.x * blockDim.x + threadIdx.x;
  if (i < n4) {
    const float4 v = ((const float4*)in)[i];
    bf16_t* o = out + (size_t)i * 4;
    o[0] = to_bf16(v.x); o[1] = to_bf16(v.y);
    o[2] = to_bf16(v.z); o[3] = to_bf16(v.w);
  }
}

// ---------------- QKV projection GEMM ----------------
// C[m,o] = sum_k X[m,k] * Wqkv[o,k]; epilogue scatters o=h*192+c into
// Q[B,H,S,D], K[B,H,S,D], Vt[B,H,D,S] (V transposed for attention B-frags).
__global__ void __launch_bounds__(256)
gemm_qkv_kernel(const bf16_t* __restrict__ X, const bf16_t* __restrict__ W,
                bf16_t* __restrict__ Qb, bf16_t* __restrict__ Kb,
                bf16_t* __restrict__ Vt) {
  __shared__ __align__(16) bf16_t smem[2 * STAGE_ELEMS];
  const int lane = threadIdx.x & 31;
  const int wave = threadIdx.x >> 5;
  const int n0 = blockIdx.x * 128;
  const int m0 = blockIdx.y * 128;
  const int wm = (wave >> 1) * 32;
  const int wn = (wave & 1) * 64;

  v8f acc[2][4] = {};
  gemm_core(X, W, m0, n0, smem, acc);

  const int col   = lane & 15;
  const int rbase = (lane >> 4) << 3;
#pragma unroll
  for (int mi = 0; mi < 2; ++mi) {
#pragma unroll
    for (int t = 0; t < 4; ++t) {
#pragma unroll
      for (int v = 0; v < 8; ++v) {
        const int m  = m0 + wm + mi * 16 + rbase + v;   // b*S + s
        const int o  = n0 + wn + t * 16 + col;          // 0..3071
        const int bi = m >> 11;
        const int s  = m & (Sdim - 1);
        const int h  = o / (3 * Dd);
        const int c  = o % (3 * Dd);
        const int d  = c & (Dd - 1);
        const bf16_t val = to_bf16(acc[mi][t][v]);
        const size_t bh = (size_t)bi * Hh + h;
        if (c < Dd)          Qb[(bh * Sdim + s) * Dd + d] = val;
        else if (c < 2 * Dd) Kb[(bh * Sdim + s) * Dd + d] = val;
        else                 Vt[(bh * Dd + d) * Sdim + s] = val;
      }
    }
  }
}

// ---------------- flash attention ----------------
__global__ void __launch_bounds__(128)
attn_kernel(const bf16_t* __restrict__ Qb, const bf16_t* __restrict__ Kb,
            const bf16_t* __restrict__ Vt, bf16_t* __restrict__ O) {
  __shared__ __align__(16) bf16_t p_lds[4][16 * 32];
  const int lane = threadIdx.x & 31;
  const int wave = threadIdx.x >> 5;
  const int bh   = blockIdx.x >> 5;
  const int q0   = (blockIdx.x & 31) * 64 + wave * 16;
  const int bi   = bh >> 4;
  const int h    = bh & 15;

  const bf16_t* Qh = Qb + (size_t)bh * Sdim * Dd;
  const bf16_t* Kh = Kb + (size_t)bh * Sdim * Dd;
  const bf16_t* Vh = Vt + (size_t)bh * Dd * Sdim;      // [D, S]

  v16bf qa0 = load_frag_a(Qh + (size_t)q0 * Dd + 0, Dd, lane);
  v16bf qa1 = load_frag_a(Qh + (size_t)q0 * Dd + 32, Dd, lane);

  v8f o_acc[4] = {};
  v8f m_run, l_run;
#pragma unroll
  for (int v = 0; v < 8; ++v) { m_run[v] = -1e30f; l_run[v] = 0.0f; }

  for (int kb = 0; kb < Sdim; kb += 32) {
    v8f s_tile[2];
#pragma unroll
    for (int t = 0; t < 2; ++t) {
      v8f c = {};
      v16bf b0 = load_frag_b(Kh + (size_t)(kb + t * 16) * Dd + 0, Dd, lane);
      c = wmma_bf16(qa0, b0, c);
      v16bf b1 = load_frag_b(Kh + (size_t)(kb + t * 16) * Dd + 32, Dd, lane);
      c = wmma_bf16(qa1, b1, c);
      s_tile[t] = c;
    }

    v8f mnew;
#pragma unroll
    for (int v = 0; v < 8; ++v) {
      float a0 = s_tile[0][v] * 0.125f;
      float a1 = s_tile[1][v] * 0.125f;
      s_tile[0][v] = a0; s_tile[1][v] = a1;
      float mx = fmaxf(a0, a1);
#pragma unroll
      for (int off = 1; off < 16; off <<= 1)
        mx = fmaxf(mx, __shfl_xor(mx, off, 32));
      mnew[v] = fmaxf(m_run[v], mx);
    }
#pragma unroll
    for (int v = 0; v < 8; ++v) {
      float p0 = __expf(s_tile[0][v] - mnew[v]);
      float p1 = __expf(s_tile[1][v] - mnew[v]);
      s_tile[0][v] = p0; s_tile[1][v] = p1;
      float sum = p0 + p1;
#pragma unroll
      for (int off = 1; off < 16; off <<= 1)
        sum += __shfl_xor(sum, off, 32);
      const float alpha = __expf(m_run[v] - mnew[v]);
      l_run[v] = l_run[v] * alpha + sum;
      m_run[v] = mnew[v];
#pragma unroll
      for (int t = 0; t < 4; ++t) o_acc[t][v] *= alpha;
    }

    const int col   = lane & 15;
    const int rbase = (lane >> 4) << 3;
    bf16_t* pl = p_lds[wave];
#pragma unroll
    for (int v = 0; v < 8; ++v) {
      pl[(rbase + v) * 32 + col]      = to_bf16(s_tile[0][v]);
      pl[(rbase + v) * 32 + 16 + col] = to_bf16(s_tile[1][v]);
    }
    v16bf pa = load_frag_a(pl, 32, lane);   // same-wave, DScnt-ordered

#pragma unroll
    for (int t = 0; t < 4; ++t) {
      v16bf vb = load_frag_b(Vh + (size_t)(t * 16) * Sdim + kb, Sdim, lane);
      o_acc[t] = wmma_bf16(pa, vb, o_acc[t]);
    }
  }

  const int col   = lane & 15;
  const int rbase = (lane >> 4) << 3;
#pragma unroll
  for (int t = 0; t < 4; ++t) {
#pragma unroll
    for (int v = 0; v < 8; ++v) {
      const int s = q0 + rbase + v;
      O[((size_t)bi * Sdim + s) * HID + h * Dd + t * 16 + col] =
          to_bf16(o_acc[t][v] / l_run[v]);
    }
  }
}

// ---------------- output projection GEMM ----------------
__global__ void __launch_bounds__(256)
gemm_out_kernel(const bf16_t* __restrict__ Xo, const bf16_t* __restrict__ Wo,
                float* __restrict__ out) {
  __shared__ __align__(16) bf16_t smem[2 * STAGE_ELEMS];
  const int lane = threadIdx.x & 31;
  const int wave = threadIdx.x >> 5;
  const int n0 = blockIdx.x * 128;
  const int m0 = blockIdx.y * 128;
  const int wm = (wave >> 1) * 32;
  const int wn = (wave & 1) * 64;

  v8f acc[2][4] = {};
  gemm_core(Xo, Wo, m0, n0, smem, acc);

  const int col   = lane & 15;
  const int rbase = (lane >> 4) << 3;
#pragma unroll
  for (int mi = 0; mi < 2; ++mi) {
#pragma unroll
    for (int t = 0; t < 4; ++t) {
#pragma unroll
      for (int v = 0; v < 8; ++v) {
        out[(size_t)(m0 + wm + mi * 16 + rbase + v) * HID +
            n0 + wn + t * 16 + col] = acc[mi][t][v];
      }
    }
  }
}

// ---------------- host launcher ----------------
extern "C" void kernel_launch(void* const* d_in, const int* in_sizes, int n_in,
                              void* d_out, int out_size, void* d_ws, size_t ws_size,
                              hipStream_t stream) {
  const float* x     = (const float*)d_in[0];
  const float* w_qkv = (const float*)d_in[1];
  const float* w_o   = (const float*)d_in[2];
  float* out = (float*)d_out;

  const size_t M    = (size_t)Bdim * Sdim;            // 8192
  const size_t nX   = M * HID;
  const size_t nWq  = (size_t)3 * HID * HID;
  const size_t nWo  = (size_t)HID * HID;
  const size_t nQKV = (size_t)Bdim * Hh * Sdim * Dd;

  char* ws = (char*)d_ws;
  size_t off = 0;
  bf16_t* Xb  = (bf16_t*)(ws + off); off += nX   * sizeof(bf16_t);
  bf16_t* Wqb = (bf16_t*)(ws + off); off += nWq  * sizeof(bf16_t);
  bf16_t* Wob = (bf16_t*)(ws + off); off += nWo  * sizeof(bf16_t);
  bf16_t* Qb  = (bf16_t*)(ws + off); off += nQKV * sizeof(bf16_t);
  bf16_t* Kb  = (bf16_t*)(ws + off); off += nQKV * sizeof(bf16_t);
  bf16_t* Vt  = (bf16_t*)(ws + off); off += nQKV * sizeof(bf16_t);
  bf16_t* Ob  = (bf16_t*)(ws + off); off += nX   * sizeof(bf16_t);
  (void)ws_size; (void)in_sizes; (void)n_in; (void)out_size;

  int n4 = (int)(nX / 4);
  cvt_bf16_kernel<<<(n4 + 255) / 256, 256, 0, stream>>>(x, Xb, n4);
  n4 = (int)(nWq / 4);
  cvt_bf16_kernel<<<(n4 + 255) / 256, 256, 0, stream>>>(w_qkv, Wqb, n4);
  n4 = (int)(nWo / 4);
  cvt_bf16_kernel<<<(n4 + 255) / 256, 256, 0, stream>>>(w_o, Wob, n4);

  gemm_qkv_kernel<<<dim3(3 * HID / 128, M / 128), 256, 0, stream>>>(
      Xb, Wqb, Qb, Kb, Vt);

  attn_kernel<<<dim3(Bdim * Hh * (Sdim / 64)), 128, 0, stream>>>(
      Qb, Kb, Vt, Ob);

  gemm_out_kernel<<<dim3(HID / 128, M / 128), 256, 0, stream>>>(Ob, Wob, out);
}
